// AttentionLayer_3590592659489
// MI455X (gfx1250) — matrix operand order repaired and checked by reading, not compiled, not run on hardware
//
#include <hip/hip_runtime.h>
#include <hip/hip_bf16.h>
#include <stdint.h>

// Problem constants (match reference)
#define ENC    1024
#define DEC    1024
#define BATCH  32
#define SEQ    2048
#define NROWS  (BATCH * SEQ)   // 65536 rows of the big GEMM
#define BM     64              // rows per workgroup in the WMMA kernel (4 M-tiles)
#define NTILES (ENC / 16)      // 64 N-tiles
#define KTILES (ENC / 32)      // 32 K-steps of 32 (bf16 WMMA K)

typedef __attribute__((ext_vector_type(16))) __bf16        v16bf;
typedef __attribute__((ext_vector_type(8)))  float         v8f;
typedef __attribute__((ext_vector_type(8)))  unsigned int  v8u;

// fp32 -> bf16 with round-to-nearest-even (bit trick)
static __device__ __forceinline__ unsigned int f32_to_bf16(float f) {
    unsigned int u = __float_as_uint(f);
    u += 0x7FFFu + ((u >> 16) & 1u);
    return u >> 16;
}

// Branchless fast tanh: tanh(x) = 1 - 2/(exp(2x)+1).
// v_exp_f32 saturates (e->0 => -1, e->inf => rcp->0 => +1), no exec divergence.
static __device__ __forceinline__ float tanh_fast(float x) {
    float e = __expf(2.0f * x);
    return 1.0f - 2.0f * __builtin_amdgcn_rcpf(e + 1.0f);
}

// ---------------------------------------------------------------------------
// Kernel 1: repack w1 (fp32 [K=1024][N=1024]) into bf16 WMMA B-fragments.
// B-fragment layout for v_wmma_f32_16x16x32_bf16 (32x16, K x N), contiguous-K order used here (same as A):
//   lane = n + 16*(k >= 16); dword j holds K = (lane>=16 ? 16 : 0) + 2j, 2j+1.
// Storage: fragment f = ntile*KTILES + kt; dword addr = (f*32 + lane)*8 + j.
// Each lane's 32 bytes are contiguous -> hot loop loads are 2x b128.
// ---------------------------------------------------------------------------
__global__ void pack_w1_bf16(const float* __restrict__ w1,
                             unsigned int* __restrict__ w1p) {
    int g    = blockIdx.x * blockDim.x + threadIdx.x;  // 0 .. 65535
    int f    = g >> 5;
    int lane = g & 31;
    int ntile = f / KTILES;
    int kt    = f % KTILES;
    int e     = ntile * 16 + (lane & 15);
    int kbase = kt * 32 + (lane >> 4) * 16;
    unsigned int* dst = w1p + (size_t)g * 8;
#pragma unroll
    for (int j = 0; j < 8; ++j) {
        int k = kbase + 2 * j;
        unsigned int lo = f32_to_bf16(w1[(size_t)k       * ENC + e]);
        unsigned int hi = f32_to_bf16(w1[(size_t)(k + 1) * ENC + e]);
        dst[j] = lo | (hi << 16);
    }
}

// ---------------------------------------------------------------------------
// Kernel 2: qq[b][e] = sum_d dec[b][d]*w2[d][e] + b1[e] + b2[e]
// (both biases folded; bv dropped — softmax is shift-invariant)
// ---------------------------------------------------------------------------
__global__ void qproj_kernel(const float* __restrict__ dec,
                             const float* __restrict__ w2,
                             const float* __restrict__ b1,
                             const float* __restrict__ b2,
                             float* __restrict__ qq) {
    __shared__ float dsh[DEC];
    int b = blockIdx.x, tid = threadIdx.x;
    for (int i = tid; i < DEC; i += 256) dsh[i] = dec[(size_t)b * DEC + i];
    __syncthreads();
    float acc[4] = {0.f, 0.f, 0.f, 0.f};
    for (int d = 0; d < DEC; ++d) {
        float dv = dsh[d];
#pragma unroll
        for (int ei = 0; ei < 4; ++ei)
            acc[ei] += dv * w2[(size_t)d * ENC + tid + ei * 256];
    }
#pragma unroll
    for (int ei = 0; ei < 4; ++ei) {
        int e = tid + ei * 256;
        qq[(size_t)b * ENC + e] = acc[ei] + b1[e] + b2[e];
    }
}

// ---------------------------------------------------------------------------
// Kernel 3 (hot): fused  score[row] = sum_n tanh( (enc@w1)[row][n] + qq[b][n] ) * v[n]
// Block = 64 threads = 2 waves. Each wave register-blocks 2 M-tiles x 2 N-tiles
// (4 concurrent WMMA accumulators) -> 2x b128 per WMMA instead of 4x.
// A tile (64 x 1024) converted fp32->bf16 and pre-swizzled into a contiguous-K
// 16-bit A-fragment layout in dynamic LDS (128 KB; 2 blocks/WGP = 1 wave/SIMD).
// ---------------------------------------------------------------------------
__global__ __launch_bounds__(64) void attn_scores_kernel(
        const float*        __restrict__ enc,
        const unsigned int* __restrict__ w1p,
        const float*        __restrict__ qq,
        const float*        __restrict__ v,
        float*              __restrict__ scores) {
    extern __shared__ unsigned int ldsA[];   // 4*KTILES*32*8 dwords = 128 KB

    const int tid     = threadIdx.x;
    const int rowbase = blockIdx.x * BM;
    const int b       = rowbase / SEQ;    // BM divides SEQ -> constant per block

    // Cooperative staging: fp32 -> bf16 -> A-fragment swizzle.
    // A layout (16x32, M x K): lanes 0-15 hold K in [0,16), lanes 16-31 hold
    // K in [16,32); dword = (K%16)/2; half = K&1.
    // A K-quad aligned to 4 lands in two CONSECUTIVE dwords of one lane segment,
    // so stage with b128 global loads + b64 LDS stores.
    for (int idx = tid; idx < BM * ENC / 4; idx += 64) {
        int m  = idx >> 8;            // ENC/4 = 256 quads per row
        int k  = (idx & 255) * 4;     // K aligned to 4
        int kt = k >> 5;
        int kk = k & 31;
        int within = kk & 15;
        int grp    = (kk >= 16) ? 1 : 0;
        int dw     = within >> 1;
        int lane   = (m & 15) + 16 * grp;
        int mt     = m >> 4;          // M-tile 0..3 within block
        float4 ev = *(const float4*)(enc + (size_t)(rowbase + m) * ENC + k);
        uint2 packed;
        packed.x = f32_to_bf16(ev.x) | (f32_to_bf16(ev.y) << 16);
        packed.y = f32_to_bf16(ev.z) | (f32_to_bf16(ev.w) << 16);
        *(uint2*)&ldsA[(((mt * KTILES + kt) * 32) + lane) * 8 + dw] = packed;
    }
    __syncthreads();

    const int wave = tid >> 5;
    const int lane = tid & 31;
    const int n_in = lane & 15;
    const int mt0  = 2 * wave;        // this wave's two M-tiles
    const int mt1  = 2 * wave + 1;

    float sacc0[8], sacc1[8];
#pragma unroll
    for (int r = 0; r < 8; ++r) { sacc0[r] = 0.f; sacc1[r] = 0.f; }

    const unsigned int* a0base = &ldsA[((mt0 * KTILES) * 32 + lane) * 8];
    const unsigned int* a1base = &ldsA[((mt1 * KTILES) * 32 + lane) * 8];

    for (int nt = 0; nt < NTILES; nt += 2) {
        v8f c00 = {}, c01 = {}, c10 = {}, c11 = {};   // c[mtile][ntile]
        const unsigned int* b0ptr = w1p + (((size_t)nt       * KTILES) * 32 + lane) * 8;
        const unsigned int* b1ptr = w1p + (((size_t)(nt + 1) * KTILES) * 32 + lane) * 8;
        if (nt + 2 < NTILES)   // pull next N-tile pair's B fragments toward L0
            __builtin_prefetch(
                w1p + (((size_t)(nt + 2) * KTILES) * 32 + lane) * 8, 0, 0);
#pragma unroll 2
        for (int kt = 0; kt < KTILES; ++kt) {
            v8u a0 = *(const v8u*)(a0base + (size_t)kt * 32 * 8);
            v8u a1 = *(const v8u*)(a1base + (size_t)kt * 32 * 8);
            v8u b0 = *(const v8u*)(b0ptr + (size_t)kt * 32 * 8);
            v8u b1 = *(const v8u*)(b1ptr + (size_t)kt * 32 * 8);
            c00 = __builtin_amdgcn_wmma_f32_16x16x32_bf16(
                    false, __builtin_bit_cast(v16bf, a0),
                    false, __builtin_bit_cast(v16bf, b0), (short)0, c00, false, false);
            c01 = __builtin_amdgcn_wmma_f32_16x16x32_bf16(
                    false, __builtin_bit_cast(v16bf, a0),
                    false, __builtin_bit_cast(v16bf, b1), (short)0, c01, false, false);
            c10 = __builtin_amdgcn_wmma_f32_16x16x32_bf16(
                    false, __builtin_bit_cast(v16bf, a1),
                    false, __builtin_bit_cast(v16bf, b0), (short)0, c10, false, false);
            c11 = __builtin_amdgcn_wmma_f32_16x16x32_bf16(
                    false, __builtin_bit_cast(v16bf, a1),
                    false, __builtin_bit_cast(v16bf, b1), (short)0, c11, false, false);
        }
        // C layout: VGPR r, lanes 0-15 -> row r, lanes 16-31 -> row 8+r; col = lane%16
        {
            int   nn = nt * 16 + n_in;
            float qn = qq[(size_t)b * ENC + nn];
            float vn = v[nn];
#pragma unroll
            for (int r = 0; r < 8; ++r) {
                sacc0[r] += tanh_fast(c00[r] + qn) * vn;
                sacc1[r] += tanh_fast(c10[r] + qn) * vn;
            }
        }
        {
            int   nn = (nt + 1) * 16 + n_in;
            float qn = qq[(size_t)b * ENC + nn];
            float vn = v[nn];
#pragma unroll
            for (int r = 0; r < 8; ++r) {
                sacc0[r] += tanh_fast(c01[r] + qn) * vn;
                sacc1[r] += tanh_fast(c11[r] + qn) * vn;
            }
        }
    }

    // Reduce across the 16 lanes of each half-wave (xor masks stay in-half)
#pragma unroll
    for (int r = 0; r < 8; ++r) {
        float s0 = sacc0[r], s1 = sacc1[r];
        s0 += __shfl_xor(s0, 1, 32);  s1 += __shfl_xor(s1, 1, 32);
        s0 += __shfl_xor(s0, 2, 32);  s1 += __shfl_xor(s1, 2, 32);
        s0 += __shfl_xor(s0, 4, 32);  s1 += __shfl_xor(s1, 4, 32);
        s0 += __shfl_xor(s0, 8, 32);  s1 += __shfl_xor(s1, 8, 32);
        sacc0[r] = s0; sacc1[r] = s1;
    }
    if ((lane & 15) == 0) {
        int mbase = (lane >> 4) * 8;   // lanes 0/16 hold rows 0-7 / 8-15
#pragma unroll
        for (int r = 0; r < 8; ++r) {
            scores[rowbase + mt0 * 16 + mbase + r] = sacc0[r];
            scores[rowbase + mt1 * 16 + mbase + r] = sacc1[r];
        }
    }
}

// ---------------------------------------------------------------------------
// Kernel 4: softmax over S=2048 per batch
// ---------------------------------------------------------------------------
__global__ void softmax_kernel(const float* __restrict__ scores,
                               float* __restrict__ attn) {
    __shared__ float red[256];
    int b = blockIdx.x, tid = threadIdx.x;
    const float* s = scores + (size_t)b * SEQ;
    float loc[8], m = -1e30f;
#pragma unroll
    for (int i = 0; i < 8; ++i) { loc[i] = s[tid + i * 256]; m = fmaxf(m, loc[i]); }
    red[tid] = m; __syncthreads();
    for (int off = 128; off > 0; off >>= 1) {
        if (tid < off) red[tid] = fmaxf(red[tid], red[tid + off]);
        __syncthreads();
    }
    m = red[0]; __syncthreads();
    float sum = 0.f;
#pragma unroll
    for (int i = 0; i < 8; ++i) { loc[i] = __expf(loc[i] - m); sum += loc[i]; }
    red[tid] = sum; __syncthreads();
    for (int off = 128; off > 0; off >>= 1) {
        if (tid < off) red[tid] += red[tid + off];
        __syncthreads();
    }
    float inv = 1.f / red[0];
    float* a = attn + (size_t)b * SEQ;
#pragma unroll
    for (int i = 0; i < 8; ++i) a[tid + i * 256] = loc[i] * inv;
}

// ---------------------------------------------------------------------------
// Kernel 5: context[b][e] = sum_s attn[b][s] * enc[b][s][e]  (streaming GEMV)
// 256 threads x float4 = full 1024-wide row, b128-coalesced per s.
// ---------------------------------------------------------------------------
__global__ void context_kernel(const float* __restrict__ enc,
                               const float* __restrict__ attn,
                               float* __restrict__ out) {
    __shared__ float ash[256];
    int b = blockIdx.x, tid = threadIdx.x;
    float4 acc = {0.f, 0.f, 0.f, 0.f};
    const float* arow = attn + (size_t)b * SEQ;
    for (int s0 = 0; s0 < SEQ; s0 += 256) {
        ash[tid] = arow[s0 + tid];
        __syncthreads();
        for (int si = 0; si < 256; ++si) {
            float a = ash[si];
            const float4 ev = *(const float4*)(
                enc + ((size_t)b * SEQ + s0 + si) * ENC + tid * 4);
            acc.x += a * ev.x; acc.y += a * ev.y;
            acc.z += a * ev.z; acc.w += a * ev.w;
        }
        __syncthreads();
    }
    *(float4*)(out + (size_t)b * ENC + tid * 4) = acc;
}

// ---------------------------------------------------------------------------
// Launch
// ---------------------------------------------------------------------------
extern "C" void kernel_launch(void* const* d_in, const int* in_sizes, int n_in,
                              void* d_out, int out_size, void* d_ws, size_t ws_size,
                              hipStream_t stream) {
    const float* enc = (const float*)d_in[0];   // [32,2048,1024]
    const float* dec = (const float*)d_in[1];   // [32,1,1024]
    const float* w1  = (const float*)d_in[2];   // [1024,1024]
    const float* b1  = (const float*)d_in[3];   // [1024]
    const float* w2  = (const float*)d_in[4];   // [1024,1024]
    const float* b2  = (const float*)d_in[5];   // [1024]
    const float* v   = (const float*)d_in[6];   // [1024,1]
    // d_in[7] = bv: constant shift, cancels in softmax.

    char* ws = (char*)d_ws;
    unsigned int* w1p    = (unsigned int*)ws;                                // 2 MB packed bf16
    float*        qq     = (float*)(ws + (size_t)(2u << 20));                // 128 KB
    float*        scores = (float*)(ws + (size_t)(2u << 20) + (128u << 10)); // 256 KB
    float*        attn   = (float*)(ws + (size_t)(2u << 20) + (384u << 10)); // 256 KB

    pack_w1_bf16      <<<(NTILES * KTILES * 32) / 256, 256, 0, stream>>>(w1, w1p);
    qproj_kernel      <<<BATCH, 256, 0, stream>>>(dec, w2, b1, b2, qq);
    attn_scores_kernel<<<NROWS / BM, 64, 4 * KTILES * 32 * 8 * sizeof(unsigned int),
                         stream>>>(enc, w1p, qq, v, scores);
    softmax_kernel    <<<BATCH, 256, 0, stream>>>(scores, attn);
    context_kernel    <<<BATCH, 256, 0, stream>>>(enc, attn, (float*)d_out);
}
